// HiLo_13494787244075
// MI455X (gfx1250) — matrix operand order repaired and checked by reading, not compiled, not run on hardware
//
#include <hip/hip_runtime.h>
#include <hip/hip_bf16.h>
#include <math.h>

typedef __bf16 bf16_t;
typedef __attribute__((ext_vector_type(8)))  __bf16 v8bf;
typedef __attribute__((ext_vector_type(16))) __bf16 v16bf;
typedef __attribute__((ext_vector_type(8)))  float  v8f;

#define C_DIM   384
#define HW_N    9216        // 96*96
#define BATCH   2
#define NTOK    (BATCH * HW_N)   // 18432
#define NHEADS  6
#define HEADD   32
#define POOLN   2304        // 48*48
#define ATT_SCALE 0.17677669529663687f   // 32^-0.5

// ---------------------------------------------------------------------------
// WMMA helpers (gfx1250, wave32). C layout per CDNA5 ISA 7.12.2; A/B use contiguous K (k = 16*(lane>>4) + i).
// ---------------------------------------------------------------------------
__device__ __forceinline__ v8f wmma_bf16(v16bf a, v16bf b, v8f c) {
    return __builtin_amdgcn_wmma_f32_16x16x32_bf16(
        /*neg_a=*/false, a, /*neg_b=*/false, b,
        /*c_mod=*/(short)0, c, /*reuse_a=*/false, /*reuse_b=*/false);
}

// A fragment (16x32 bf16): lane holds row m = lane&15; 16 sequential K
// starting at (lane>>4)*16.  Two b128 loads.
__device__ __forceinline__ v16bf load_a(const bf16_t* p, int ld, int lane) {
    const bf16_t* q = p + (size_t)(lane & 15) * ld + ((lane >> 4) << 4);
    v8bf lo = *(const v8bf*)q;
    v8bf hi = *(const v8bf*)(q + 8);
    v16bf r;
#pragma unroll
    for (int i = 0; i < 8; ++i) { r[i] = lo[i]; r[i + 8] = hi[i]; }
    return r;
}

// B fragment (32x16 bf16, stored as B^T row-major [n, K]): lane holds column
// n = lane&15; 16 sequential K starting at (lane>>4)*16.  Two b128 loads.
__device__ __forceinline__ v16bf load_b(const bf16_t* p, int ld, int lane) {
    const bf16_t* q = p + (size_t)(lane & 15) * ld + ((lane >> 4) << 4);
    v8bf lo = *(const v8bf*)q;
    v8bf hi = *(const v8bf*)(q + 8);
    v16bf r;
#pragma unroll
    for (int i = 0; i < 8; ++i) { r[i] = lo[i]; r[i + 8] = hi[i]; }
    return r;
}

// ---------------------------------------------------------------------------
// Layout / conversion kernels
// ---------------------------------------------------------------------------
// x [B,C,HW] f32 -> Xbf [B,HW,C] bf16 via 32x32 LDS-tiled transpose:
// coalesced f32 reads along s, coalesced bf16 writes along c.
__global__ void k_cvt_x(const float* __restrict__ x, bf16_t* __restrict__ Xbf) {
    __shared__ float t[32][33];
    int bs = blockIdx.x * 32;          // spatial base (HW_N/32 = 288)
    int bc = blockIdx.y * 32;          // channel base (C_DIM/32 = 12)
    int b  = blockIdx.z;
    int tx = threadIdx.x & 31;
    int ty = threadIdx.x >> 5;         // 0..7
    const float* xp = x + ((size_t)b * C_DIM + bc) * HW_N + bs;
#pragma unroll
    for (int i = 0; i < 4; ++i) {
        int c = ty + i * 8;
        t[c][tx] = xp[(size_t)c * HW_N + tx];
    }
    __syncthreads();
    bf16_t* op = Xbf + ((size_t)b * HW_N + bs) * C_DIM + bc;
#pragma unroll
    for (int i = 0; i < 4; ++i) {
        int s = ty + i * 8;
        op[(size_t)s * C_DIM + tx] = (bf16_t)t[tx][s];
    }
}

// W [K,N] f32 -> Wt [N,K] bf16
__global__ void k_cvt_wT(const float* __restrict__ W, bf16_t* __restrict__ Wt,
                         int K, int N) {
    int i = blockIdx.x * blockDim.x + threadIdx.x;
    if (i >= K * N) return;
    int n = i % N, k = i / N;
    Wt[(size_t)n * K + k] = (bf16_t)W[i];
}

// 2x2 avg-pool on x [B,C,H,W] -> Xp [B, POOLN, C] bf16
__global__ void k_pool(const float* __restrict__ x, bf16_t* __restrict__ Xp) {
    int i = blockIdx.x * blockDim.x + threadIdx.x;
    if (i >= BATCH * C_DIM * POOLN) return;
    int p = i % POOLN;
    int c = (i / POOLN) % C_DIM;
    int b = i / (POOLN * C_DIM);
    int py = p / 48, px = p % 48;
    const float* base = x + ((size_t)(b * C_DIM + c)) * HW_N + (py * 2) * 96 + px * 2;
    float v = 0.25f * (base[0] + base[1] + base[96] + base[97]);
    Xp[((size_t)(b * POOLN + p)) * C_DIM + c] = (bf16_t)v;
}

// Ykv [B*POOLN, 384] bf16 -> K [B,h,M,32] and V^T [B,h,32,M]
__global__ void k_repack_kv(const bf16_t* __restrict__ Ykv,
                            bf16_t* __restrict__ Kb, bf16_t* __restrict__ Vt) {
    int i = blockIdx.x * blockDim.x + threadIdx.x;
    if (i >= BATCH * POOLN * 384) return;
    int col = i % 384;
    int m   = i / 384;
    int b = m / POOLN, mm = m % POOLN;
    int sel = col / 192;
    int h = (col % 192) / 32;
    int d = col % 32;
    bf16_t v = Ykv[i];
    if (sel == 0)
        Kb[(((size_t)(b * NHEADS + h) * POOLN) + mm) * 32 + d] = v;
    else
        Vt[(((size_t)(b * NHEADS + h) * 32) + d) * POOLN + mm] = v;
}

// ---------------------------------------------------------------------------
// Tiled WMMA GEMM.  C[M,N] = A[M,K] x Bt[N,K]^T.
// One wave = 16x64 tile (1 A-fragment shared by 4 B-fragments / 4 WMMAs per
// K-step).  N must be a multiple of 64; K a multiple of 32.
// mode 0: write bf16 row-major to Cb.
// mode 1: write fp32 + bias, transposed into BCHW output at channel c0+n.
// ---------------------------------------------------------------------------
__global__ void k_gemm(const bf16_t* __restrict__ A, const bf16_t* __restrict__ Bt,
                       bf16_t* __restrict__ Cb, float* __restrict__ outBCHW,
                       const float* __restrict__ bias,
                       int M, int N, int K, int mode, int c0) {
    int lane = threadIdx.x & 31;
    int wid  = blockIdx.x * (blockDim.x >> 5) + (threadIdx.x >> 5);
    int tilesN = N >> 6;                       // N/64
    int tiles  = (M >> 4) * tilesN;
    if (wid >= tiles) return;
    int tm = wid / tilesN, tn = wid % tilesN;
    const bf16_t* Ap = A  + (size_t)tm * 16 * K;
    const bf16_t* Bp = Bt + (size_t)(tn * 64) * K;
    v8f acc[4];
#pragma unroll
    for (int j = 0; j < 4; ++j)
#pragma unroll
        for (int r = 0; r < 8; ++r) acc[j][r] = 0.f;
    for (int k = 0; k < K; k += 32) {
        if (k + 64 < K) {
            __builtin_prefetch(Ap + k + 64, 0, 1);
            __builtin_prefetch(Bp + k + 64, 0, 1);
        }
        v16bf a = load_a(Ap + k, K, lane);
#pragma unroll
        for (int j = 0; j < 4; ++j) {
            v16bf b = load_b(Bp + (size_t)(j * 16) * K + k, K, lane);
            acc[j] = wmma_bf16(a, b, acc[j]);
        }
    }
    int mbase = tm * 16 + ((lane >> 4) << 3);
    if (mode == 0) {
#pragma unroll
        for (int j = 0; j < 4; ++j) {
            int ncol = tn * 64 + j * 16 + (lane & 15);
#pragma unroll
            for (int r = 0; r < 8; ++r)
                Cb[(size_t)(mbase + r) * N + ncol] = (bf16_t)acc[j][r];
        }
    } else {
#pragma unroll
        for (int j = 0; j < 4; ++j) {
            int ncol = tn * 64 + j * 16 + (lane & 15);
            float bv = bias[ncol];
            int ch = c0 + ncol;
#pragma unroll
            for (int r = 0; r < 8; ++r) {
                int m = mbase + r;
                int b = m / HW_N, s = m % HW_N;
                outBCHW[((size_t)b * C_DIM + ch) * HW_N + s] = acc[j][r] + bv;
            }
        }
    }
}

// ---------------------------------------------------------------------------
// Hi-Fi: local attention over 2x2 windows (n=4) — scalar VALU, tiny compute.
// One thread per (b, head, window, query-token).
// ---------------------------------------------------------------------------
__global__ void k_hifi(const bf16_t* __restrict__ Y1, bf16_t* __restrict__ hiA) {
    int t = blockIdx.x * blockDim.x + threadIdx.x;
    const int total = BATCH * NHEADS * POOLN * 4;
    if (t >= total) return;
    int qi  = t & 3;
    int wxy = (t >> 2) % POOLN;
    int h   = ((t >> 2) / POOLN) % NHEADS;
    int b   = t / (4 * POOLN * NHEADS);
    int wy = wxy / 48, wx = wxy % 48;
    int tok[4];
#pragma unroll
    for (int j = 0; j < 4; ++j)
        tok[j] = (wy * 2 + (j >> 1)) * 96 + (wx * 2 + (j & 1));
    const bf16_t* base = Y1 + (size_t)b * HW_N * 576;
    float q[32];
    const bf16_t* qp = base + (size_t)tok[qi] * 576 + h * 32;
#pragma unroll
    for (int d = 0; d < 32; ++d) q[d] = (float)qp[d];
    float sc[4];
    float mx = -1e30f;
#pragma unroll
    for (int j = 0; j < 4; ++j) {
        const bf16_t* kp = base + (size_t)tok[j] * 576 + 192 + h * 32;
        float s = 0.f;
#pragma unroll
        for (int d = 0; d < 32; ++d) s += q[d] * (float)kp[d];
        sc[j] = s * ATT_SCALE;
        mx = fmaxf(mx, sc[j]);
    }
    float sum = 0.f;
#pragma unroll
    for (int j = 0; j < 4; ++j) { sc[j] = __expf(sc[j] - mx); sum += sc[j]; }
    float inv = 1.f / sum;
    float acc[32];
#pragma unroll
    for (int d = 0; d < 32; ++d) acc[d] = 0.f;
#pragma unroll
    for (int j = 0; j < 4; ++j) {
        const bf16_t* vp = base + (size_t)tok[j] * 576 + 384 + h * 32;
        float p = sc[j] * inv;
#pragma unroll
        for (int d = 0; d < 32; ++d) acc[d] += p * (float)vp[d];
    }
    bf16_t* op = hiA + ((size_t)(b * HW_N + tok[qi])) * 192 + h * 32;
#pragma unroll
    for (int d = 0; d < 32; ++d) op[d] = (bf16_t)acc[d];
}

// ---------------------------------------------------------------------------
// Lo-Fi: flash attention, one wave per 16-query tile per (b, head).
// Scores via WMMA (q 16x32 x K^T 32x16), online softmax through LDS,
// P·V via WMMA against V^T [d, M].
// ---------------------------------------------------------------------------
__global__ void k_lofi(const bf16_t* __restrict__ Yq, const bf16_t* __restrict__ Kb,
                       const bf16_t* __restrict__ Vt, bf16_t* __restrict__ loA) {
    __shared__ __align__(16) float  sS[16 * 32];
    __shared__ __align__(16) bf16_t sP[16 * 32];
    __shared__ float sFac[16];
    int lane = threadIdx.x;
    int blk  = blockIdx.x;                 // B * NHEADS * (HW_N/16)
    const int QT = HW_N / 16;
    int qt = blk % QT;
    int h  = (blk / QT) % NHEADS;
    int b  = blk / (QT * NHEADS);

    const bf16_t* Qp = Yq + ((size_t)(b * HW_N + qt * 16)) * 192 + h * 32;
    v16bf qa = load_a(Qp, 192, lane);
    const bf16_t* Kp = Kb + (size_t)(b * NHEADS + h) * POOLN * 32;
    const bf16_t* Vp = Vt + (size_t)(b * NHEADS + h) * 32 * POOLN;

    v8f acc0 = {0.f, 0.f, 0.f, 0.f, 0.f, 0.f, 0.f, 0.f};
    v8f acc1 = {0.f, 0.f, 0.f, 0.f, 0.f, 0.f, 0.f, 0.f};
    float mrow = -1e30f, lrow = 0.f;       // row stats live in lanes 0..15
    int nn = lane & 15, mb = (lane >> 4) << 3;

    for (int kv = 0; kv < POOLN; kv += 32) {
        v16bf kb0 = load_b(Kp + (size_t)kv * 32, 32, lane);
        v16bf kb1 = load_b(Kp + (size_t)(kv + 16) * 32, 32, lane);
        v8f z = {0.f, 0.f, 0.f, 0.f, 0.f, 0.f, 0.f, 0.f};
        v8f s0 = wmma_bf16(qa, kb0, z);
        v8f s1 = wmma_bf16(qa, kb1, z);
#pragma unroll
        for (int r = 0; r < 8; ++r) {
            sS[(mb + r) * 32 + nn]      = s0[r] * ATT_SCALE;
            sS[(mb + r) * 32 + 16 + nn] = s1[r] * ATT_SCALE;
        }
        __syncthreads();
        if (lane < 16) {
            float mx = mrow;
#pragma unroll
            for (int j = 0; j < 32; ++j) mx = fmaxf(mx, sS[lane * 32 + j]);
            float sum = 0.f;
#pragma unroll
            for (int j = 0; j < 32; ++j) {
                float p = __expf(sS[lane * 32 + j] - mx);
                sP[lane * 32 + j] = (bf16_t)p;
                sum += p;
            }
            float fac = __expf(mrow - mx);
            lrow = lrow * fac + sum;
            mrow = mx;
            sFac[lane] = fac;
        }
        __syncthreads();
#pragma unroll
        for (int r = 0; r < 8; ++r) {
            float f = sFac[mb + r];
            acc0[r] *= f;
            acc1[r] *= f;
        }
        v16bf pa = load_a(sP, 32, lane);
        v16bf v0 = load_b(Vp + kv, POOLN, lane);
        v16bf v1 = load_b(Vp + (size_t)16 * POOLN + kv, POOLN, lane);
        acc0 = wmma_bf16(pa, v0, acc0);
        acc1 = wmma_bf16(pa, v1, acc1);
        __syncthreads();
    }
    if (lane < 16) sFac[lane] = lrow;
    __syncthreads();
    bf16_t* Op = loA + ((size_t)(b * HW_N + qt * 16)) * 192 + h * 32;
#pragma unroll
    for (int r = 0; r < 8; ++r) {
        float inv = 1.f / sFac[mb + r];
        Op[(size_t)(mb + r) * 192 + nn]      = (bf16_t)(acc0[r] * inv);
        Op[(size_t)(mb + r) * 192 + 16 + nn] = (bf16_t)(acc1[r] * inv);
    }
}

// ---------------------------------------------------------------------------
extern "C" void kernel_launch(void* const* d_in, const int* in_sizes, int n_in,
                              void* d_out, int out_size, void* d_ws, size_t ws_size,
                              hipStream_t stream) {
    (void)in_sizes; (void)n_in; (void)out_size; (void)ws_size;
    const float* x       = (const float*)d_in[0];
    const float* Wh_qkv  = (const float*)d_in[1];
    const float* Wh_proj = (const float*)d_in[2];
    const float* bh_proj = (const float*)d_in[3];
    const float* Wl_q    = (const float*)d_in[4];
    const float* Wl_kv   = (const float*)d_in[5];
    const float* Wl_proj = (const float*)d_in[6];
    const float* bl_proj = (const float*)d_in[7];
    float* out = (float*)d_out;

    char* ws = (char*)d_ws;
    size_t off = 0;
    auto alloc = [&](size_t bytes) {
        void* p = ws + off;
        off = (off + bytes + 255) & ~(size_t)255;
        return p;
    };
    bf16_t* Xbf     = (bf16_t*)alloc((size_t)NTOK * C_DIM * 2);
    bf16_t* WhqkvT  = (bf16_t*)alloc((size_t)576 * 384 * 2);
    bf16_t* WlqT    = (bf16_t*)alloc((size_t)192 * 384 * 2);
    bf16_t* WlkvT   = (bf16_t*)alloc((size_t)384 * 384 * 2);
    bf16_t* WhprojT = (bf16_t*)alloc((size_t)192 * 192 * 2);
    bf16_t* WlprojT = (bf16_t*)alloc((size_t)192 * 192 * 2);
    bf16_t* Y1      = (bf16_t*)alloc((size_t)NTOK * 576 * 2);
    bf16_t* Yq      = (bf16_t*)alloc((size_t)NTOK * 192 * 2);
    bf16_t* Xp      = (bf16_t*)alloc((size_t)BATCH * POOLN * 384 * 2);
    bf16_t* Ykv     = (bf16_t*)alloc((size_t)BATCH * POOLN * 384 * 2);
    bf16_t* Kbuf    = (bf16_t*)alloc((size_t)BATCH * NHEADS * POOLN * 32 * 2);
    bf16_t* Vtb     = (bf16_t*)alloc((size_t)BATCH * NHEADS * 32 * POOLN * 2);
    bf16_t* hiA     = (bf16_t*)alloc((size_t)NTOK * 192 * 2);
    bf16_t* loA     = (bf16_t*)alloc((size_t)NTOK * 192 * 2);

    // conversions
    dim3 tgrid(HW_N / 32, C_DIM / 32, BATCH);
    k_cvt_x<<<tgrid, 256, 0, stream>>>(x, Xbf);
    k_cvt_wT<<<(384 * 576 + 255) / 256, 256, 0, stream>>>(Wh_qkv, WhqkvT, 384, 576);
    k_cvt_wT<<<(384 * 192 + 255) / 256, 256, 0, stream>>>(Wl_q, WlqT, 384, 192);
    k_cvt_wT<<<(384 * 384 + 255) / 256, 256, 0, stream>>>(Wl_kv, WlkvT, 384, 384);
    k_cvt_wT<<<(192 * 192 + 255) / 256, 256, 0, stream>>>(Wh_proj, WhprojT, 192, 192);
    k_cvt_wT<<<(192 * 192 + 255) / 256, 256, 0, stream>>>(Wl_proj, WlprojT, 192, 192);
    int np = BATCH * C_DIM * POOLN;
    k_pool<<<(np + 255) / 256, 256, 0, stream>>>(x, Xp);

    auto gemm_blocks = [](int M, int N) {
        int tiles = (M / 16) * (N / 64);
        return (tiles * 32 + 255) / 256;
    };
    // input GEMMs
    k_gemm<<<gemm_blocks(NTOK, 576), 256, 0, stream>>>(
        Xbf, WhqkvT, Y1, nullptr, nullptr, NTOK, 576, 384, 0, 0);
    k_gemm<<<gemm_blocks(NTOK, 192), 256, 0, stream>>>(
        Xbf, WlqT, Yq, nullptr, nullptr, NTOK, 192, 384, 0, 0);
    k_gemm<<<gemm_blocks(BATCH * POOLN, 384), 256, 0, stream>>>(
        Xp, WlkvT, Ykv, nullptr, nullptr, BATCH * POOLN, 384, 384, 0, 0);
    k_repack_kv<<<(BATCH * POOLN * 384 + 255) / 256, 256, 0, stream>>>(Ykv, Kbuf, Vtb);

    // attention
    k_hifi<<<(BATCH * NHEADS * POOLN * 4 + 127) / 128, 128, 0, stream>>>(Y1, hiA);
    k_lofi<<<BATCH * NHEADS * (HW_N / 16), 32, 0, stream>>>(Yq, Kbuf, Vtb, loA);

    // projection GEMMs, fused bias + transposed store into BCHW output
    k_gemm<<<gemm_blocks(NTOK, 192), 256, 0, stream>>>(
        hiA, WhprojT, nullptr, out, bh_proj, NTOK, 192, 192, 1, 0);
    k_gemm<<<gemm_blocks(NTOK, 192), 256, 0, stream>>>(
        loA, WlprojT, nullptr, out, bl_proj, NTOK, 192, 192, 1, 192);
}
